// DenseDSnetwork_12352325943902
// MI455X (gfx1250) — matrix operand order repaired and checked by reading, not compiled, not run on hardware
//
#include <hip/hip_runtime.h>
#include <hip/hip_bf16.h>
#include <math.h>

typedef __bf16 bf16_t;
typedef bf16_t v8bf  __attribute__((ext_vector_type(8)));
typedef bf16_t v16bf __attribute__((ext_vector_type(16)));
typedef float  v8f   __attribute__((ext_vector_type(8)));

#define S_TOT 131072
#define G_TOT 4096
#define D_DIM 256
#define H_DIM 512
#define T_DIM 10
#define L_NUM 3

// ---------------- fragment loaders (wave32 WMMA; contiguous k order, same for A and B) ---------
// A (16x32 bf16, M x K): lanes 0-15 row M=lane, K [k0..k0+15];
// lanes 16-31 row M=lane-16, K [k0+16..k0+31].
__device__ __forceinline__ v16bf load_fragA(const bf16_t* A, int ld, int row0, int k0) {
    const int lane = threadIdx.x & 31;
    const bf16_t* p = A + (size_t)(row0 + (lane & 15)) * ld + (k0 + ((lane >> 4) << 4));
    v8bf lo = *reinterpret_cast<const v8bf*>(p);        // 16B -> global_load_b128
    v8bf hi = *reinterpret_cast<const v8bf*>(p + 8);    // 16B -> global_load_b128
    return __builtin_shufflevector(lo, hi, 0,1,2,3,4,5,6,7,8,9,10,11,12,13,14,15);
}

// B (32x16 bf16, K x N) supplied as Bt[N][K] row-major: lane&15 selects column N,
// lane>=16 takes the upper K half (16 contiguous bf16 = 32B load).
__device__ __forceinline__ v16bf load_fragB(const bf16_t* Bt, int ld, int n0, int k0) {
    const int lane = threadIdx.x & 31;
    const bf16_t* p = Bt + (size_t)(n0 + (lane & 15)) * ld + (k0 + ((lane >> 4) << 4));
    return *reinterpret_cast<const v16bf*>(p);
}

// ---------------- WMMA GEMM: C[M x N] = A[M x K] * Bt^T, block = 64 rows x 256 cols ---
// MODE 0: outF = acc + bias1[col] + bias2[col]                    (x2 = m@fcs_w + biases)
// MODE 1: outB = elu(acc + x2[idx[row]][col])  (in-place h update; barrier before write)
// MODE 2: outF = relu(acc + bias1[col])                           (hidden layer)
template<int MODE>
__global__ __launch_bounds__(256) void gemm_wmma_kernel(
    const bf16_t* A, const bf16_t* Bt, int K, int ldc,
    float* outF, bf16_t* outB,
    const float* __restrict__ bias1, const float* __restrict__ bias2,
    const int* __restrict__ idx, const float* __restrict__ x2)
{
    const int w    = threadIdx.x >> 5;
    const int lane = threadIdx.x & 31;
    const int wm   = w >> 2;                       // 0..1  (32-row sub-blocks)
    const int wn   = w & 3;                        // 0..3  (64-col sub-blocks)
    const int rowBase = blockIdx.x * 64 + wm * 32;
    const int colBase = blockIdx.y * 256 + wn * 64;

    v8f zero = {};
    v8f acc[2][4];
    #pragma unroll
    for (int i = 0; i < 2; ++i)
        #pragma unroll
        for (int j = 0; j < 4; ++j) acc[i][j] = zero;

    for (int k0 = 0; k0 < K; k0 += 32) {
        v16bf a0 = load_fragA(A, K, rowBase,      k0);
        v16bf a1 = load_fragA(A, K, rowBase + 16, k0);
        v16bf b0 = load_fragB(Bt, K, colBase,      k0);
        v16bf b1 = load_fragB(Bt, K, colBase + 16, k0);
        v16bf b2 = load_fragB(Bt, K, colBase + 32, k0);
        v16bf b3 = load_fragB(Bt, K, colBase + 48, k0);
        acc[0][0] = __builtin_amdgcn_wmma_f32_16x16x32_bf16(false, a0, false, b0, (short)0, acc[0][0], false, false);
        acc[0][1] = __builtin_amdgcn_wmma_f32_16x16x32_bf16(false, a0, false, b1, (short)0, acc[0][1], false, false);
        acc[0][2] = __builtin_amdgcn_wmma_f32_16x16x32_bf16(false, a0, false, b2, (short)0, acc[0][2], false, false);
        acc[0][3] = __builtin_amdgcn_wmma_f32_16x16x32_bf16(false, a0, false, b3, (short)0, acc[0][3], false, false);
        acc[1][0] = __builtin_amdgcn_wmma_f32_16x16x32_bf16(false, a1, false, b0, (short)0, acc[1][0], false, false);
        acc[1][1] = __builtin_amdgcn_wmma_f32_16x16x32_bf16(false, a1, false, b1, (short)0, acc[1][1], false, false);
        acc[1][2] = __builtin_amdgcn_wmma_f32_16x16x32_bf16(false, a1, false, b2, (short)0, acc[1][2], false, false);
        acc[1][3] = __builtin_amdgcn_wmma_f32_16x16x32_bf16(false, a1, false, b3, (short)0, acc[1][3], false, false);
    }

    // All A reads are complete before any wave writes (required for MODE 1 in-place).
    __syncthreads();

    // C/D layout: VGPR j -> row j (lanes 0-15) / j+8 (lanes 16-31); col = lane&15.
    const int colLane = lane & 15;
    const int rowOff  = (lane >> 4) * 8;
    #pragma unroll
    for (int i = 0; i < 2; ++i) {
        #pragma unroll
        for (int jj = 0; jj < 8; ++jj) {
            const int row = rowBase + i * 16 + rowOff + jj;
            if (MODE == 1) {
                const int g = idx[row];
                const float* xr = x2 + (size_t)g * D_DIM;
                #pragma unroll
                for (int j4 = 0; j4 < 4; ++j4) {
                    const int col = colBase + j4 * 16 + colLane;
                    float v = acc[i][j4][jj] + xr[col];
                    v = (v > 0.0f) ? v : (__expf(v) - 1.0f);   // ELU(alpha=1)
                    outB[(size_t)row * ldc + col] = (bf16_t)v;
                }
            } else if (MODE == 0) {
                #pragma unroll
                for (int j4 = 0; j4 < 4; ++j4) {
                    const int col = colBase + j4 * 16 + colLane;
                    outF[(size_t)row * ldc + col] = acc[i][j4][jj] + bias1[col] + bias2[col];
                }
            } else {
                #pragma unroll
                for (int j4 = 0; j4 < 4; ++j4) {
                    const int col = colBase + j4 * 16 + colLane;
                    float v = acc[i][j4][jj] + bias1[col];
                    outF[(size_t)row * ldc + col] = fmaxf(v, 0.0f);
                }
            }
        }
    }
}

// ---------------- helper kernels -----------------------------------------------------
__global__ void convert_f32_to_bf16(const float* __restrict__ in, bf16_t* __restrict__ out, int n) {
    int t = blockIdx.x * blockDim.x + threadIdx.x;
    for (; t < n; t += gridDim.x * blockDim.x) out[t] = (bf16_t)in[t];
}

// Wt[n*K + k] = (bf16) W[k*Nin + n]
__global__ void transpose_to_bf16(const float* __restrict__ W, bf16_t* __restrict__ Wt, int K, int Nin) {
    int t = blockIdx.x * blockDim.x + threadIdx.x;
    if (t >= K * Nin) return;
    int n = t / K;
    int k = t - n * K;
    Wt[t] = (bf16_t)W[k * Nin + n];
}

__global__ void count_kernel(const int* __restrict__ idx, int* __restrict__ cnt, int n) {
    int t = blockIdx.x * blockDim.x + threadIdx.x;
    if (t < n) atomicAdd(&cnt[idx[t]], 1);
}

// Segment-sum over a sorted index: run-length accumulation per 128-row strip.
// One thread per column (blockDim.x == D_DIM).
__global__ __launch_bounds__(256) void segsum_kernel(const bf16_t* __restrict__ h,
                                                     const int* __restrict__ idx,
                                                     float* __restrict__ msum, int rowsPerBlock) {
    const int col = threadIdx.x;
    const int r0 = blockIdx.x * rowsPerBlock;
    const int r1 = r0 + rowsPerBlock;
    float acc = 0.0f;
    int cur = idx[r0];
    for (int r = r0; r < r1; ++r) {
        int g = idx[r];
        if (g != cur) {
            atomicAdd(&msum[(size_t)cur * D_DIM + col], acc);
            acc = 0.0f; cur = g;
        }
        acc += (float)h[(size_t)r * D_DIM + col];
    }
    atomicAdd(&msum[(size_t)cur * D_DIM + col], acc);
}

__global__ void normalize_kernel(const float* __restrict__ msum, const int* __restrict__ cnt,
                                 bf16_t* __restrict__ out, int n) {
    int t = blockIdx.x * blockDim.x + threadIdx.x;
    if (t >= n) return;
    int g = t >> 8;  // /D_DIM
    int c = cnt[g];
    float denom = (float)(c > 1 ? c : 1);
    out[t] = (bf16_t)(msum[t] / denom);
}

// out[g][t] = hidden[g] . f2w[:,t] + f2b[t]
__global__ __launch_bounds__(512) void final_fc_kernel(const float* __restrict__ hidden,
                                                       const float* __restrict__ f2w,
                                                       const float* __restrict__ f2b,
                                                       float* __restrict__ out) {
    __shared__ float row[H_DIM];
    const int g = blockIdx.x;
    row[threadIdx.x] = hidden[(size_t)g * H_DIM + threadIdx.x];
    __syncthreads();
    if (threadIdx.x < T_DIM) {
        float s = f2b[threadIdx.x];
        #pragma unroll 8
        for (int k = 0; k < H_DIM; ++k) s += row[k] * f2w[k * T_DIM + threadIdx.x];
        out[g * T_DIM + threadIdx.x] = s;
    }
}

// ---------------- host orchestration -------------------------------------------------
extern "C" void kernel_launch(void* const* d_in, const int* in_sizes, int n_in,
                              void* d_out, int out_size, void* d_ws, size_t ws_size,
                              hipStream_t stream) {
    (void)in_sizes; (void)n_in; (void)out_size;
    const float* h0    = (const float*)d_in[0];   // [S, D]
    const int*   idx   = (const int*)d_in[1];     // [S]
    const float* fc_w  = (const float*)d_in[2];   // [L, D, D]
    const float* fc_b  = (const float*)d_in[3];   // [L, D]
    const float* fcs_w = (const float*)d_in[4];   // [L, D, D]
    const float* fcs_b = (const float*)d_in[5];   // [L, D]
    const float* f1_w  = (const float*)d_in[6];   // [D, H]
    const float* f1_b  = (const float*)d_in[7];   // [H]
    const float* f2_w  = (const float*)d_in[8];   // [H, T]
    const float* f2_b  = (const float*)d_in[9];   // [T]
    float* out = (float*)d_out;                   // [G, T]

    // workspace carve-out (256B aligned)
    unsigned char* p = (unsigned char*)d_ws;
    auto alloc = [&](size_t bytes) -> void* {
        void* r = (void*)p;
        p += (bytes + 255) & ~(size_t)255;
        return r;
    };
    bf16_t* hbf    = (bf16_t*)alloc((size_t)S_TOT * D_DIM * 2);   // 64 MB, in-place activations
    float*  msum   = (float*) alloc((size_t)G_TOT * D_DIM * 4);
    bf16_t* mbf    = (bf16_t*)alloc((size_t)G_TOT * D_DIM * 2);
    float*  x2     = (float*) alloc((size_t)G_TOT * D_DIM * 4);
    int*    cnt    = (int*)   alloc((size_t)G_TOT * 4);
    bf16_t* WtA    = (bf16_t*)alloc((size_t)D_DIM * D_DIM * 2);
    bf16_t* WtB    = (bf16_t*)alloc((size_t)D_DIM * D_DIM * 2);
    bf16_t* Wt1    = (bf16_t*)alloc((size_t)H_DIM * D_DIM * 2);
    bf16_t* hgbf   = (bf16_t*)alloc((size_t)G_TOT * D_DIM * 2);
    float*  hidden = (float*) alloc((size_t)G_TOT * H_DIM * 4);
    if ((size_t)(p - (unsigned char*)d_ws) > ws_size) return;     // ws too small: bail

    // prologue: h -> bf16; counts (idx is static across layers)
    convert_f32_to_bf16<<<2048, 256, 0, stream>>>(h0, hbf, S_TOT * D_DIM);
    hipMemsetAsync(cnt, 0, (size_t)G_TOT * 4, stream);
    count_kernel<<<(S_TOT + 255) / 256, 256, 0, stream>>>(idx, cnt, S_TOT);

    for (int i = 0; i < L_NUM; ++i) {
        // m = seg_mean(h)
        hipMemsetAsync(msum, 0, (size_t)G_TOT * D_DIM * 4, stream);
        segsum_kernel<<<S_TOT / 128, 256, 0, stream>>>(hbf, idx, msum, 128);
        normalize_kernel<<<(G_TOT * D_DIM) / 256, 256, 0, stream>>>(msum, cnt, mbf, G_TOT * D_DIM);
        // weight prep (transpose + bf16)
        transpose_to_bf16<<<(D_DIM * D_DIM + 255) / 256, 256, 0, stream>>>(fc_w  + (size_t)i * D_DIM * D_DIM, WtA, D_DIM, D_DIM);
        transpose_to_bf16<<<(D_DIM * D_DIM + 255) / 256, 256, 0, stream>>>(fcs_w + (size_t)i * D_DIM * D_DIM, WtB, D_DIM, D_DIM);
        // x2 = m @ fcs_w + fcs_b + fc_b   [G x D]
        gemm_wmma_kernel<0><<<dim3(G_TOT / 64, 1), 256, 0, stream>>>(
            mbf, WtB, D_DIM, D_DIM, x2, nullptr,
            fcs_b + (size_t)i * D_DIM, fc_b + (size_t)i * D_DIM, nullptr, nullptr);
        // h = elu(h @ fc_w + x2[idx])  (in-place, bf16)   [S x D]
        gemm_wmma_kernel<1><<<dim3(S_TOT / 64, 1), 256, 0, stream>>>(
            hbf, WtA, D_DIM, D_DIM, nullptr, hbf,
            nullptr, nullptr, idx, x2);
    }

    // h_graph = seg_mean(h)
    hipMemsetAsync(msum, 0, (size_t)G_TOT * D_DIM * 4, stream);
    segsum_kernel<<<S_TOT / 128, 256, 0, stream>>>(hbf, idx, msum, 128);
    normalize_kernel<<<(G_TOT * D_DIM) / 256, 256, 0, stream>>>(msum, cnt, hgbf, G_TOT * D_DIM);

    // hidden = relu(h_graph @ f1_w + f1_b)   [G x H]
    transpose_to_bf16<<<(D_DIM * H_DIM + 255) / 256, 256, 0, stream>>>(f1_w, Wt1, D_DIM, H_DIM);
    gemm_wmma_kernel<2><<<dim3(G_TOT / 64, H_DIM / 256), 256, 0, stream>>>(
        hgbf, Wt1, D_DIM, H_DIM, hidden, nullptr, f1_b, nullptr, nullptr, nullptr);

    // out = hidden @ f2_w + f2_b   [G x T]
    final_fc_kernel<<<G_TOT, H_DIM, 0, stream>>>(hidden, f2_w, f2_b, out);
}
